// AttentionLayer_89223650607328
// MI455X (gfx1250) — matrix operand order repaired and checked by reading, not compiled, not run on hardware
//
#include <hip/hip_runtime.h>
#include <hip/hip_bf16.h>
#include <stdint.h>

#define DIM   512
#define KD    1024
#define BB    4
#define TT    128
#define SS    512

typedef __attribute__((ext_vector_type(16))) _Float16 v16h;
typedef __attribute__((ext_vector_type(8)))  _Float16 v8h;
typedef __attribute__((ext_vector_type(8)))  float    v8f;

// ---------------------------------------------------------------------------
// Flat f32 -> f16 convert (feeds WMMA A-operands)
// ---------------------------------------------------------------------------
__global__ void cvt_f32_to_f16(const float* __restrict__ in, _Float16* __restrict__ out, int n) {
  int i = blockIdx.x * blockDim.x + threadIdx.x;
  if (i < n) out[i] = (_Float16)in[i];
}

// in: [K][N] f32 row-major, out: [N][K] f16 (transposed so B-fragments are
// contiguous 16B loads along K). Tiny matrices; traffic negligible.
__global__ void transpose_cvt(const float* __restrict__ in, _Float16* __restrict__ out,
                              int K, int N) {
  int k = blockIdx.x * blockDim.x + threadIdx.x;
  int n = blockIdx.y;
  if (k < K) out[(size_t)n * K + k] = (_Float16)in[(size_t)k * N + n];
}

// ---------------------------------------------------------------------------
// WMMA fragment loaders (CDNA5 ISA 7.12.2 lane layouts):
//   A 16x32 f16 : lane L -> row M=L&15; half h=L>>4 -> K = {h*8..h*8+7} (v0-3)
//                 and {16+h*8..+7} (v4-7)  => two contiguous b128 loads
//   B 32x16 f16 : lane L -> col N=L&15; half h -> K = {h*8..h*8+7} (v0-3)
//                 and {16+h*8..+7} (v4-7) => two b128 loads from the transposed weight
// ---------------------------------------------------------------------------
__device__ __forceinline__ v16h load_frag_a(const _Float16* __restrict__ row,
                                            int kb, int kh) {
  v8h a0 = *(const v8h*)(row + kb + kh * 8);
  v8h a1 = *(const v8h*)(row + kb + 16 + kh * 8);
  v16h a;
#pragma unroll
  for (int j = 0; j < 8; ++j) { a[j] = a0[j]; a[8 + j] = a1[j]; }
  return a;
}

__device__ __forceinline__ v16h load_frag_b(const _Float16* __restrict__ row,
                                            int kb, int kh) {
  v8h b0 = *(const v8h*)(row + kb + kh * 8);
  v8h b1 = *(const v8h*)(row + kb + 16 + kh * 8);
  v16h b;
#pragma unroll
  for (int j = 0; j < 8; ++j) { b[j] = b0[j]; b[8 + j] = b1[j]; }
  return b;
}

#define WMMA_F16(a, b, c) \
  __builtin_amdgcn_wmma_f32_16x16x32_f16(false, (a), false, (b), (short)0, (c), false, false)

// ---------------------------------------------------------------------------
// One wave computes a 32x32 macro-tile of C = A(f16,MxK) * Bt(f16,NxK)^T + bias
// with 2x2 register blocking: 2 A-frags + 2 B-frags -> 4 WMMAs per k-step of 32
// (8 b128 loads : 4 v_wmma, double the reuse of the 1-tile version).
// C/D 16x16 f32 layout: vgpr r, lane L -> (M = r + 8*(L>>4), N = L&15).
// ---------------------------------------------------------------------------
__global__ __launch_bounds__(32) void gemm32_wmma(
    const _Float16* __restrict__ A, const _Float16* __restrict__ Bt,
    const float* __restrict__ bias, float* __restrict__ C,
    int M, int N, int K) {
  const int lane = threadIdx.x & 31;
  const int kh   = lane >> 4;   // half selector
  const int l16  = lane & 15;
  const int nBase = blockIdx.x * 32;
  const int mBase = blockIdx.y * 32;

  const _Float16* arow0 = A  + (size_t)(mBase + l16) * K;
  const _Float16* arow1 = arow0 + (size_t)16 * K;
  const _Float16* brow0 = Bt + (size_t)(nBase + l16) * K;
  const _Float16* brow1 = brow0 + (size_t)16 * K;

  v8f acc00 = {}, acc01 = {}, acc10 = {}, acc11 = {};
  for (int kb = 0; kb < K; kb += 32) {
    if (kb + 32 < K) {
      // WGP-scope prefetch of the next k-slab (fragments are re-read by the
      // other waves sharing this row/column of tiles on the same WGP)
      __builtin_prefetch(arow0 + kb + 32, 0, 3);
      __builtin_prefetch(arow1 + kb + 32, 0, 3);
      __builtin_prefetch(brow0 + kb + 32, 0, 3);
      __builtin_prefetch(brow1 + kb + 32, 0, 3);
    }
    v16h a0 = load_frag_a(arow0, kb, kh);
    v16h a1 = load_frag_a(arow1, kb, kh);
    v16h b0 = load_frag_b(brow0, kb, kh);
    v16h b1 = load_frag_b(brow1, kb, kh);
    acc00 = WMMA_F16(a0, b0, acc00);
    acc01 = WMMA_F16(a0, b1, acc01);
    acc10 = WMMA_F16(a1, b0, acc10);
    acc11 = WMMA_F16(a1, b1, acc11);
  }

  const int n0 = nBase + l16;
  const int n1 = n0 + 16;
  const float bias0 = bias[n0];
  const float bias1 = bias[n1];
#pragma unroll
  for (int r = 0; r < 8; ++r) {
    const int m0 = mBase + r + kh * 8;
    const int m1 = m0 + 16;
    C[(size_t)m0 * N + n0] = acc00[r] + bias0;
    C[(size_t)m0 * N + n1] = acc01[r] + bias1;
    C[(size_t)m1 * N + n0] = acc10[r] + bias0;
    C[(size_t)m1 * N + n1] = acc11[r] + bias1;
  }
}

// ---------------------------------------------------------------------------
// Fused: additive scores (tanh) + softmax + P*V per (b,t) row.
// Never materializes the (B,T,S,D) tensor (would be 536 MB).
// ---------------------------------------------------------------------------
#if __has_builtin(__builtin_amdgcn_tanhf)
__device__ __forceinline__ float fast_tanh(float x) { return __builtin_amdgcn_tanhf(x); }
#else
__device__ __forceinline__ float fast_tanh(float x) {
  float e = __expf(-2.0f * fabsf(x));
  float r = (1.0f - e) / (1.0f + e);
  return copysignf(r, x);
}
#endif

__global__ __launch_bounds__(256) void attn_fused(
    const float* __restrict__ Kp, const float* __restrict__ Qp,
    const float* __restrict__ V,  const uint8_t* __restrict__ mask,
    const float* __restrict__ wi, const float* __restrict__ bi,
    float* __restrict__ outAttn,  float* __restrict__ outScores) {
  __shared__ float qrow[DIM];
  __shared__ float wis[DIM];
  __shared__ float sc[SS];
  __shared__ float red[256];

  const int b = blockIdx.y, t = blockIdx.x;
  const int tid = threadIdx.x;

  for (int d = tid; d < DIM; d += 256) {
    qrow[d] = Qp[((size_t)(b * TT + t)) * DIM + d];
    wis[d]  = wi[d];
  }
  __syncthreads();

  // --- scores: 8 waves stride over S, lanes split D (coalesced, bank-clean) ---
  const int wave = tid >> 5, lane = tid & 31;
  const float biv = bi[0];
  for (int s = wave; s < SS; s += 8) {
    const float* krow = Kp + ((size_t)(b * SS + s)) * DIM;
    float acc = 0.0f;
#pragma unroll 4
    for (int i = 0; i < DIM / 32; ++i) {
      const int d = lane + i * 32;
      acc += fast_tanh(krow[d] + qrow[d]) * wis[d];
    }
#pragma unroll
    for (int off = 16; off > 0; off >>= 1) acc += __shfl_xor(acc, off, 32);
    if (lane == 0) sc[s] = mask[b * SS + s] ? -INFINITY : (acc + biv);
  }
  __syncthreads();

  // --- softmax over sc[0..SS) ---
  float m = -INFINITY;
  for (int s = tid; s < SS; s += 256) m = fmaxf(m, sc[s]);
  red[tid] = m; __syncthreads();
  for (int o = 128; o > 0; o >>= 1) {
    if (tid < o) red[tid] = fmaxf(red[tid], red[tid + o]);
    __syncthreads();
  }
  const float mx = red[0];
  __syncthreads();

  float sum = 0.0f;
  for (int s = tid; s < SS; s += 256) {
    const float e = __expf(sc[s] - mx);
    sc[s] = e; sum += e;
  }
  red[tid] = sum; __syncthreads();
  for (int o = 128; o > 0; o >>= 1) {
    if (tid < o) red[tid] += red[tid + o];
    __syncthreads();
  }
  const float inv = 1.0f / red[0];
  __syncthreads();

  for (int s = tid; s < SS; s += 256) {
    const float p = sc[s] * inv;
    sc[s] = p;
    outScores[((size_t)(b * TT + t)) * SS + s] = p;  // (B,T,1,S) flat
  }
  __syncthreads();

  // --- attn row = P * V (V rows are L2-resident; coalesced over d) ---
  for (int d = tid; d < DIM; d += 256) {
    float acc = 0.0f;
#pragma unroll 8
    for (int s = 0; s < SS; ++s)
      acc += sc[s] * V[((size_t)(b * SS + s)) * DIM + d];
    outAttn[((size_t)(b * TT + t)) * DIM + d] = acc;
  }
}

// ---------------------------------------------------------------------------
extern "C" void kernel_launch(void* const* d_in, const int* in_sizes, int n_in,
                              void* d_out, int out_size, void* d_ws, size_t ws_size,
                              hipStream_t stream) {
  const float*   query = (const float*)d_in[0];
  const float*   key   = (const float*)d_in[1];
  const float*   value = (const float*)d_in[2];
  const uint8_t* mask  = (const uint8_t*)d_in[3];
  const float*   Wk    = (const float*)d_in[4];
  const float*   bk    = (const float*)d_in[5];
  const float*   Wq    = (const float*)d_in[6];
  const float*   bq    = (const float*)d_in[7];
  const float*   wi    = (const float*)d_in[8];
  const float*   bi    = (const float*)d_in[9];

  float* outAttn   = (float*)d_out;
  float* outScores = outAttn + (size_t)BB * TT * DIM;

  char* ws = (char*)d_ws;
  _Float16* keyH = (_Float16*)ws;  ws += (size_t)BB * SS * KD  * sizeof(_Float16);
  _Float16* qryH = (_Float16*)ws;  ws += (size_t)BB * TT * DIM * sizeof(_Float16);
  _Float16* WkT  = (_Float16*)ws;  ws += (size_t)KD  * DIM * sizeof(_Float16);
  _Float16* WqT  = (_Float16*)ws;  ws += (size_t)DIM * DIM * sizeof(_Float16);
  float*    Kp   = (float*)ws;     ws += (size_t)BB * SS * DIM * sizeof(float);
  float*    Qp   = (float*)ws;     ws += (size_t)BB * TT * DIM * sizeof(float);

  { const int n = BB * SS * KD;
    cvt_f32_to_f16<<<(n + 255) / 256, 256, 0, stream>>>(key, keyH, n); }
  { const int n = BB * TT * DIM;
    cvt_f32_to_f16<<<(n + 255) / 256, 256, 0, stream>>>(query, qryH, n); }
  transpose_cvt<<<dim3((KD  + 255) / 256, DIM), 256, 0, stream>>>(Wk, WkT, KD,  DIM);
  transpose_cvt<<<dim3((DIM + 255) / 256, DIM), 256, 0, stream>>>(Wq, WqT, DIM, DIM);

  // key_proj: (B*S=2048, K=1024) x (1024, 512), 32x32 tiles
  gemm32_wmma<<<dim3(DIM / 32, (BB * SS) / 32), 32, 0, stream>>>(
      keyH, WkT, bk, Kp, BB * SS, DIM, KD);
  // query_proj: (B*T=512, K=512) x (512, 512), 32x32 tiles
  gemm32_wmma<<<dim3(DIM / 32, (BB * TT) / 32), 32, 0, stream>>>(
      qryH, WqT, bq, Qp, BB * TT, DIM, DIM);

  attn_fused<<<dim3(TT, BB), 256, 0, stream>>>(
      Kp, Qp, value, mask, wi, bi, outAttn, outScores);
}
